// GeoAligner_70523363000370
// MI455X (gfx1250) — matrix operand order repaired and checked by reading, not compiled, not run on hardware
//
#include <hip/hip_runtime.h>
#include <math.h>

typedef __attribute__((ext_vector_type(16))) _Float16 v16h;
typedef __attribute__((ext_vector_type(8)))  _Float16 v8h;
typedef __attribute__((ext_vector_type(8)))  float    v8f;

#define BDIM 128   // batch (both modalities)
#define NTOK 128   // image tokens
#define MTOK 32    // gps tokens
#define DDIM 256   // feature dim

#define INV_EPS   20.0f          // 1 / 0.05
#define LOG_MU1  (-4.852030263919617f)   // -log(128)
#define LOG_NU1  (-3.4657359027997265f)  // -log(32)
#define LOG_MU2  (-3.4657359027997265f)  // -log(32)
#define LOG_NU2  (-4.852030263919617f)   // -log(128)
#define HYB_L    0.2f

// ---------------------------------------------------------------------------
// L2-normalize rows of (rows x 256) f32, emit f16. One block (256 thr) per row.
// ---------------------------------------------------------------------------
__global__ void l2norm_to_f16_kernel(const float* __restrict__ in,
                                     _Float16* __restrict__ out) {
  __shared__ float red[256];
  const int row = blockIdx.x;
  const int t   = threadIdx.x;
  const float x = in[(size_t)row * DDIM + t];
  red[t] = x * x;
  __syncthreads();
  for (int s = 128; s > 0; s >>= 1) {
    if (t < s) red[t] += red[t + s];
    __syncthreads();
  }
  const float inv = 1.0f / fmaxf(sqrtf(red[0]), 1e-12f);
  out[(size_t)row * DDIM + t] = (_Float16)(x * inv);
}

__device__ __forceinline__ float block_reduce_sum256(float v, float* red, int t) {
  red[t] = v;
  __syncthreads();
  for (int s = 128; s > 0; s >>= 1) {
    if (t < s) red[t] += red[t + s];
    __syncthreads();
  }
  const float r = red[0];
  __syncthreads();
  return r;
}

// ---------------------------------------------------------------------------
// One block per (b = img batch, g = gps batch) pair.
// Phase 1: S(128x32) = imgH_b(128x256) * gpsH_g^T via v_wmma_f32_16x16x32_f16.
// Phase 2: hybrid late/Sinkhorn scores for both directions from the one tile.
// ---------------------------------------------------------------------------
__global__ void __launch_bounds__(256)
pair_score_kernel(const _Float16* __restrict__ imgH,
                  const _Float16* __restrict__ gpsH,
                  float* __restrict__ Li2g,   // [B,B]  (img-batch major)
                  float* __restrict__ Lg2i) { // [B,B]  (gps-batch major)
  __shared__ float S[NTOK * MTOK];     // 16 KB similarity tile
  __shared__ float lu[NTOK], lv[MTOK];
  __shared__ float lu2[MTOK], lv2[NTOK];
  __shared__ float pmax[256], psum[256];
  __shared__ float red[256];

  const int b    = blockIdx.x;   // image batch index
  const int g    = blockIdx.y;   // gps batch index
  const int t    = threadIdx.x;
  const int lane = t & 31;
  const int wave = t >> 5;       // 8 waves -> wave owns rows [wave*16, wave*16+16)
  const int half = lane >> 4;    // 0 or 1 (lane half)
  const int l16  = lane & 15;

  // ---------------- Phase 1: WMMA GEMM ----------------
  {
    const _Float16* Arow  = imgH + ((size_t)b * NTOK + wave * 16 + l16) * DDIM;
    const _Float16* Grow0 = gpsH + ((size_t)g * MTOK + l16) * DDIM;        // N-tile 0
    const _Float16* Grow1 = Grow0 + 16 * DDIM;                             // N-tile 1
    const int cA = half * 16;   // A: K-base 0 or 16
    const int cB = half * 16;   // B: K-base 0 or 16

    v8f acc0 = {}; v8f acc1 = {};
    for (int kk = 0; kk < DDIM; kk += 32) {
      const v8h a_lo = *(const v8h*)(Arow + kk + cA);
      const v8h a_hi = *(const v8h*)(Arow + kk + cA + 8);
      const v8h b0lo = *(const v8h*)(Grow0 + kk + cB);
      const v8h b0hi = *(const v8h*)(Grow0 + kk + cB + 8);
      const v8h b1lo = *(const v8h*)(Grow1 + kk + cB);
      const v8h b1hi = *(const v8h*)(Grow1 + kk + cB + 8);
      v16h a, bf0, bf1;
#pragma unroll
      for (int i = 0; i < 8; ++i) {
        a[i] = a_lo[i];  a[8 + i] = a_hi[i];
        bf0[i] = b0lo[i]; bf0[8 + i] = b0hi[i];
        bf1[i] = b1lo[i]; bf1[8 + i] = b1hi[i];
      }
      acc0 = __builtin_amdgcn_wmma_f32_16x16x32_f16(false, a, false, bf0,
                                                    (short)0, acc0, false, false);
      acc1 = __builtin_amdgcn_wmma_f32_16x16x32_f16(false, a, false, bf1,
                                                    (short)0, acc1, false, false);
    }
    // Scatter D tiles into LDS per C/D layout: M = r + 8*half, N = l16.
    const int mbase = wave * 16 + half * 8;
#pragma unroll
    for (int r = 0; r < 8; ++r) {
      S[(mbase + r) * MTOK + l16]      = acc0[r];
      S[(mbase + r) * MTOK + l16 + 16] = acc1[r];
    }
  }
  __syncthreads();

  const int cm  = t & 31;   // column id for 8-segment column passes
  const int seg = t >> 5;   // segment id (16 rows each)

  // ---------------- late scores ----------------
  float contribA = 0.f;
  if (t < NTOK) {
    float rmax = -3.0e38f;
    for (int m = 0; m < MTOK; ++m) rmax = fmaxf(rmax, S[t * MTOK + m]);
    contribA = rmax;
  }
  const float lateA = block_reduce_sum256(contribA, red, t) * (1.0f / NTOK);

  {
    float cmax = -3.0e38f;
    for (int r = 0; r < 16; ++r) cmax = fmaxf(cmax, S[(seg * 16 + r) * MTOK + cm]);
    pmax[t] = cmax;
  }
  __syncthreads();
  float contribB = 0.f;
  if (t < MTOK) {
    float mm = pmax[t];
    for (int s = 1; s < 8; ++s) mm = fmaxf(mm, pmax[s * 32 + t]);
    contribB = mm;
  }
  __syncthreads();
  const float lateB = block_reduce_sum256(contribB, red, t) * (1.0f / MTOK);

  // ---------------- Sinkhorn direction 1: rows = img tokens (N x M) ----------
  if (t < MTOK) lv[t] = 0.f;
  __syncthreads();
  for (int it = 0; it < 3; ++it) {
    if (t < NTOK) {
      float mx = -3.0e38f;
      for (int m = 0; m < MTOK; ++m)
        mx = fmaxf(mx, S[t * MTOK + m] * INV_EPS + lv[m]);
      float sm = 0.f;
      for (int m = 0; m < MTOK; ++m)
        sm += __expf(S[t * MTOK + m] * INV_EPS + lv[m] - mx);
      lu[t] = LOG_MU1 - (mx + __logf(sm));
    }
    __syncthreads();
    {
      float mx = -3.0e38f;
      for (int r = 0; r < 16; ++r) {
        const int n = seg * 16 + r;
        mx = fmaxf(mx, S[n * MTOK + cm] * INV_EPS + lu[n]);
      }
      float sm = 0.f;
      for (int r = 0; r < 16; ++r) {
        const int n = seg * 16 + r;
        sm += __expf(S[n * MTOK + cm] * INV_EPS + lu[n] - mx);
      }
      pmax[t] = mx; psum[t] = sm;
    }
    __syncthreads();
    if (t < MTOK) {
      float mm = -3.0e38f;
      for (int s = 0; s < 8; ++s) mm = fmaxf(mm, pmax[s * 32 + t]);
      float ss = 0.f;
      for (int s = 0; s < 8; ++s) ss += psum[s * 32 + t] * __expf(pmax[s * 32 + t] - mm);
      lv[t] = LOG_NU1 - (mm + __logf(ss));
    }
    __syncthreads();
  }
  float acc = 0.f;
#pragma unroll 4
  for (int i = 0; i < 16; ++i) {
    const int e = t * 16 + i;
    const int n = e >> 5, m = e & 31;
    const float s = S[e];
    acc += __expf(s * INV_EPS + lu[n] + lv[m]) * s;
  }
  const float ot1 = block_reduce_sum256(acc, red, t);

  // ---------------- Sinkhorn direction 2: rows = gps tokens (M x N) ----------
  if (t < NTOK) lv2[t] = 0.f;
  __syncthreads();
  for (int it = 0; it < 3; ++it) {
    {
      float mx = -3.0e38f;
      for (int r = 0; r < 16; ++r) {
        const int n = seg * 16 + r;
        mx = fmaxf(mx, S[n * MTOK + cm] * INV_EPS + lv2[n]);
      }
      float sm = 0.f;
      for (int r = 0; r < 16; ++r) {
        const int n = seg * 16 + r;
        sm += __expf(S[n * MTOK + cm] * INV_EPS + lv2[n] - mx);
      }
      pmax[t] = mx; psum[t] = sm;
    }
    __syncthreads();
    if (t < MTOK) {
      float mm = -3.0e38f;
      for (int s = 0; s < 8; ++s) mm = fmaxf(mm, pmax[s * 32 + t]);
      float ss = 0.f;
      for (int s = 0; s < 8; ++s) ss += psum[s * 32 + t] * __expf(pmax[s * 32 + t] - mm);
      lu2[t] = LOG_MU2 - (mm + __logf(ss));
    }
    __syncthreads();
    if (t < NTOK) {
      float mx = -3.0e38f;
      for (int m = 0; m < MTOK; ++m)
        mx = fmaxf(mx, S[t * MTOK + m] * INV_EPS + lu2[m]);
      float sm = 0.f;
      for (int m = 0; m < MTOK; ++m)
        sm += __expf(S[t * MTOK + m] * INV_EPS + lu2[m] - mx);
      lv2[t] = LOG_NU2 - (mx + __logf(sm));
    }
    __syncthreads();
  }
  float acc2 = 0.f;
#pragma unroll 4
  for (int i = 0; i < 16; ++i) {
    const int e = t * 16 + i;
    const int n = e >> 5, m = e & 31;
    const float s = S[e];
    acc2 += __expf(s * INV_EPS + lu2[m] + lv2[n]) * s;
  }
  const float ot2 = block_reduce_sum256(acc2, red, t);

  if (t == 0) {
    Li2g[b * BDIM + g] = (1.0f - HYB_L) * lateA + HYB_L * ot1;
    Lg2i[g * BDIM + b] = (1.0f - HYB_L) * lateB + HYB_L * ot2;
  }
}

// ---------------------------------------------------------------------------
// Final scalar loss: scaled log-softmax diagonal mean of both logit matrices.
// ---------------------------------------------------------------------------
__global__ void loss_kernel(const float* __restrict__ Li2g,
                            const float* __restrict__ Lg2i,
                            const float* __restrict__ logit_scale,
                            float* __restrict__ out) {
  __shared__ float red[128];
  const int t = threadIdx.x;  // 128 threads, one row of each matrix
  const float scale = fminf(__expf(logit_scale[0]), 100.0f);
  float c = 0.f;
  {
    float mx = -3.0e38f;
    for (int j = 0; j < BDIM; ++j) mx = fmaxf(mx, Li2g[t * BDIM + j] * scale);
    float sm = 0.f;
    for (int j = 0; j < BDIM; ++j) sm += __expf(Li2g[t * BDIM + j] * scale - mx);
    c += Li2g[t * BDIM + t] * scale - (mx + __logf(sm));
  }
  {
    float mx = -3.0e38f;
    for (int j = 0; j < BDIM; ++j) mx = fmaxf(mx, Lg2i[t * BDIM + j] * scale);
    float sm = 0.f;
    for (int j = 0; j < BDIM; ++j) sm += __expf(Lg2i[t * BDIM + j] * scale - mx);
    c += Lg2i[t * BDIM + t] * scale - (mx + __logf(sm));
  }
  red[t] = c;
  __syncthreads();
  for (int s = 64; s > 0; s >>= 1) {
    if (t < s) red[t] += red[t + s];
    __syncthreads();
  }
  if (t == 0) out[0] = -red[0] / (2.0f * BDIM);
}

extern "C" void kernel_launch(void* const* d_in, const int* in_sizes, int n_in,
                              void* d_out, int out_size, void* d_ws, size_t ws_size,
                              hipStream_t stream) {
  (void)in_sizes; (void)n_in; (void)out_size; (void)ws_size;
  const float* img    = (const float*)d_in[0];   // (128,128,256) f32
  const float* gps    = (const float*)d_in[1];   // (128,32,256)  f32
  const float* lscale = (const float*)d_in[2];   // scalar f32
  float* out = (float*)d_out;

  char* ws = (char*)d_ws;
  const size_t imgH_bytes = (size_t)BDIM * NTOK * DDIM * sizeof(_Float16); // 8 MB
  const size_t gpsH_bytes = (size_t)BDIM * MTOK * DDIM * sizeof(_Float16); // 2 MB
  _Float16* imgH = (_Float16*)ws;
  _Float16* gpsH = (_Float16*)(ws + imgH_bytes);
  float*    Li2g = (float*)(ws + imgH_bytes + gpsH_bytes);
  float*    Lg2i = Li2g + BDIM * BDIM;

  l2norm_to_f16_kernel<<<BDIM * NTOK, 256, 0, stream>>>(img, imgH);
  l2norm_to_f16_kernel<<<BDIM * MTOK, 256, 0, stream>>>(gps, gpsH);
  pair_score_kernel<<<dim3(BDIM, BDIM), 256, 0, stream>>>(imgH, gpsH, Li2g, Lg2i);
  loss_kernel<<<1, 128, 0, stream>>>(Li2g, Lg2i, lscale, out);
}
